// SelfAttention_1030792151308
// MI455X (gfx1250) — matrix operand order repaired and checked by reading, not compiled, not run on hardware
//
#include <hip/hip_runtime.h>

// Self-attention forward returning (out, attn): fp32 in, bf16 WMMA compute.
// B=2 H=12 N=2048 D=64. d_out = out[B,H,N,D] ++ attn[B,H,N,N] (fp32).

typedef __attribute__((ext_vector_type(16))) __bf16 v16bf;
typedef __attribute__((ext_vector_type(4)))  __bf16 v4bf;
typedef __attribute__((ext_vector_type(8)))  float  v8f;

constexpr int Bb = 2;
constexpr int Hh = 12;
constexpr int Nn = 2048;
constexpr int Dd = 64;
constexpr int BM = 128;           // rows per workgroup (8 waves x 16 rows)
constexpr int BN = 64;            // key tile
constexpr int VT_PAD = 2;         // LDS bank-conflict padding for V^T
constexpr float NEGV  = -1e10f;
constexpr float SCALE = 0.125f;   // 1/sqrt(64)

__device__ __forceinline__ v8f wmma_bf16(v16bf a, v16bf b, v8f c) {
  return __builtin_amdgcn_wmma_f32_16x16x32_bf16(
      /*neg_a=*/false, a, /*neg_b=*/false, b,
      /*c_mod=*/(short)0, c, /*reuse_a=*/false, /*reuse_b=*/false);
}

// B-matrix fragment: 16 consecutive K-dim bf16 values for this lane's column.
__device__ __forceinline__ v16bf load_bfrag(const __bf16* p) {
  v16bf r;
#pragma unroll
  for (int i = 0; i < 16; ++i) r[i] = p[i];
  return r;
}

// A-matrix fragment: lane (m, half) of a 16x32 tile taken from a 64-wide row.
// elems 0..7  -> K = 32*kb + 16*half + (0..7)
// elems 8..15 -> K = 32*kb + 16*half + 8 + (0..7)
__device__ __forceinline__ v16bf load_afrag(const __bf16* row, int kb, int half) {
  v16bf r;
  const __bf16* c0 = row + kb * 32 + half * 16;
  const __bf16* c1 = row + kb * 32 + half * 16 + 8;
#pragma unroll
  for (int i = 0; i < 8; ++i) { r[i] = c0[i]; r[8 + i] = c1[i]; }
  return r;
}

__global__ __launch_bounds__(256)
void attn_fwd_kernel(const float* __restrict__ q, const float* __restrict__ k,
                     const float* __restrict__ v,
                     const unsigned char* __restrict__ mask,
                     float* __restrict__ out, float* __restrict__ attn) {
  // q-staging and P-staging never live simultaneously: share 16 KB.
  __shared__ union {
    __bf16 q[BM][Dd];
    __bf16 p[8][16][BN];
  } qp;
  __shared__ __bf16 k_lds[BN][Dd];             // 8 KB
  __shared__ __bf16 vt_lds[Dd][BN + VT_PAD];   // ~8.25 KB, V transposed
  __shared__ unsigned int mbits[BM][2];        // 1 KB packed mask tile

  const int tid  = threadIdx.x;
  const int wave = tid >> 5;
  const int lane = tid & 31;
  const int half = lane >> 4;
  const int l16  = lane & 15;

  const int bh   = blockIdx.x >> 4;            // N/BM == 16 blocks per (b,h)
  const int mblk = blockIdx.x & 15;
  const int row0 = mblk * BM;

  const float* qb = q + (size_t)bh * Nn * Dd;
  const float* kb = k + (size_t)bh * Nn * Dd;
  const float* vb = v + (size_t)bh * Nn * Dd;
  const unsigned char* mb = mask + (size_t)bh * Nn * Nn;
  float* ob = out  + (size_t)bh * Nn * Dd;
  float* ab = attn + (size_t)bh * Nn * Nn;

  // per-lane mask-bit selectors (column l16 within the two 16-col halves of a word)
  const unsigned bsel0 = 1u << l16;
  const unsigned bsel1 = 1u << (16 + l16);

  // mask staging coords: thread -> (row, 32-col word)
  const int mrl = tid >> 1;            // 0..127
  const int mcw = tid & 1;             // 0..1

  // prefetch coords: thread -> (row 0..63, 64B chunk 0..3)
  const int pfr = tid >> 2;
  const int pfc = (tid & 3) * 16;

  // ---- stage Q block (fp32 -> bf16, vectorized), pull per-wave A frags ----
  for (int i = tid; i < (BM * Dd) / 4; i += 256) {
    const int r = i >> 4, c4 = (i & 15) * 4;
    const float4 f = *(const float4*)&qb[(size_t)(row0 + r) * Dd + c4];
    v4bf h; h[0] = (__bf16)f.x; h[1] = (__bf16)f.y; h[2] = (__bf16)f.z; h[3] = (__bf16)f.w;
    *(v4bf*)&qp.q[r][c4] = h;
  }
  __syncthreads();
  const v16bf qa0 = load_afrag(&qp.q[wave * 16 + l16][0], 0, half);
  const v16bf qa1 = load_afrag(&qp.q[wave * 16 + l16][0], 1, half);

  float mrow[8], lrow[8];
#pragma unroll
  for (int r = 0; r < 8; ++r) { mrow[r] = -__builtin_inff(); lrow[r] = 0.f; }

  // =================== Pass 1: row max + denominator ===================
  for (int n0 = 0; n0 < Nn; n0 += BN) {
    __syncthreads();
    if (n0 + BN < Nn)   // prefetch next K tile (global_prefetch_b8)
      __builtin_prefetch(&kb[(size_t)(n0 + BN + pfr) * Dd + pfc], 0, 1);

    // stage K tile (vectorized fp32 -> bf16)
    for (int i = tid; i < (BN * Dd) / 4; i += 256) {
      const int r = i >> 4, c4 = (i & 15) * 4;
      const float4 f = *(const float4*)&kb[(size_t)(n0 + r) * Dd + c4];
      v4bf h; h[0] = (__bf16)f.x; h[1] = (__bf16)f.y; h[2] = (__bf16)f.z; h[3] = (__bf16)f.w;
      *(v4bf*)&k_lds[r][c4] = h;
    }
    // stage mask tile as packed bits: 32 bytes -> one u32 per thread
    {
      const unsigned char* mp = mb + (size_t)(row0 + mrl) * Nn + n0 + mcw * 32;
      const uint4 wa = *(const uint4*)(mp);
      const uint4 wb2 = *(const uint4*)(mp + 16);
      const unsigned wrds[8] = {wa.x, wa.y, wa.z, wa.w, wb2.x, wb2.y, wb2.z, wb2.w};
      unsigned bits = 0;
#pragma unroll
      for (int wd = 0; wd < 8; ++wd) {
        // collapse each byte's nonzero-ness into its bit0
        unsigned t = wrds[wd];
        t |= t >> 4; t |= t >> 2; t |= t >> 1;
        t &= 0x01010101u;
        bits |= (t & 1u) << (wd * 4 + 0);
        bits |= ((t >> 8) & 1u) << (wd * 4 + 1);
        bits |= ((t >> 16) & 1u) << (wd * 4 + 2);
        bits |= ((t >> 24) & 1u) << (wd * 4 + 3);
      }
      mbits[mrl][mcw] = bits;
    }
    __syncthreads();

    v8f s[4];
#pragma unroll
    for (int nt = 0; nt < 4; ++nt) {
      const int key = nt * 16 + l16;
      v16bf b0 = load_bfrag(&k_lds[key][half * 16]);
      v16bf b1 = load_bfrag(&k_lds[key][32 + half * 16]);
      v8f acc = {};
      acc = wmma_bf16(qa0, b0, acc);
      acc = wmma_bf16(qa1, b1, acc);
      s[nt] = acc;
    }
    // scale + mask from packed bits (LDS broadcast reads)
#pragma unroll
    for (int r = 0; r < 8; ++r) {
      const int rl = wave * 16 + r + 8 * half;
      const unsigned w0 = mbits[rl][0], w1 = mbits[rl][1];
      s[0][r] = (w0 & bsel0) ? NEGV : s[0][r] * SCALE;
      s[1][r] = (w0 & bsel1) ? NEGV : s[1][r] * SCALE;
      s[2][r] = (w1 & bsel0) ? NEGV : s[2][r] * SCALE;
      s[3][r] = (w1 & bsel1) ? NEGV : s[3][r] * SCALE;
    }

#pragma unroll
    for (int r = 0; r < 8; ++r) {
      float t = fmaxf(fmaxf(s[0][r], s[1][r]), fmaxf(s[2][r], s[3][r]));
      t = fmaxf(t, __shfl_xor(t, 1, 32));
      t = fmaxf(t, __shfl_xor(t, 2, 32));
      t = fmaxf(t, __shfl_xor(t, 4, 32));
      t = fmaxf(t, __shfl_xor(t, 8, 32));
      const float mnew = fmaxf(mrow[r], t);
      float es = __expf(s[0][r] - mnew) + __expf(s[1][r] - mnew) +
                 __expf(s[2][r] - mnew) + __expf(s[3][r] - mnew);
      es += __shfl_xor(es, 1, 32);
      es += __shfl_xor(es, 2, 32);
      es += __shfl_xor(es, 4, 32);
      es += __shfl_xor(es, 8, 32);
      lrow[r] = lrow[r] * __expf(mrow[r] - mnew) + es;
      mrow[r] = mnew;
    }
  }

  float inv_l[8];
#pragma unroll
  for (int r = 0; r < 8; ++r) inv_l[r] = 1.f / lrow[r];

  // ============ Pass 2: write normalized attn, compute O = P*V ============
  v8f o[4];
#pragma unroll
  for (int dt = 0; dt < 4; ++dt) o[dt] = (v8f){};

  for (int n0 = 0; n0 < Nn; n0 += BN) {
    __syncthreads();
    if (n0 + BN < Nn) {  // prefetch next K and V tiles
      __builtin_prefetch(&kb[(size_t)(n0 + BN + pfr) * Dd + pfc], 0, 1);
      __builtin_prefetch(&vb[(size_t)(n0 + BN + pfr) * Dd + pfc], 0, 1);
    }
    for (int i = tid; i < (BN * Dd) / 4; i += 256) {
      const int r = i >> 4, c4 = (i & 15) * 4;
      const float4 fk = *(const float4*)&kb[(size_t)(n0 + r) * Dd + c4];
      v4bf h; h[0] = (__bf16)fk.x; h[1] = (__bf16)fk.y; h[2] = (__bf16)fk.z; h[3] = (__bf16)fk.w;
      *(v4bf*)&k_lds[r][c4] = h;
      const float4 fv = *(const float4*)&vb[(size_t)(n0 + r) * Dd + c4];
      vt_lds[c4 + 0][r] = (__bf16)fv.x;
      vt_lds[c4 + 1][r] = (__bf16)fv.y;
      vt_lds[c4 + 2][r] = (__bf16)fv.z;
      vt_lds[c4 + 3][r] = (__bf16)fv.w;
    }
    {
      const unsigned char* mp = mb + (size_t)(row0 + mrl) * Nn + n0 + mcw * 32;
      const uint4 wa = *(const uint4*)(mp);
      const uint4 wb2 = *(const uint4*)(mp + 16);
      const unsigned wrds[8] = {wa.x, wa.y, wa.z, wa.w, wb2.x, wb2.y, wb2.z, wb2.w};
      unsigned bits = 0;
#pragma unroll
      for (int wd = 0; wd < 8; ++wd) {
        unsigned t = wrds[wd];
        t |= t >> 4; t |= t >> 2; t |= t >> 1;
        t &= 0x01010101u;
        bits |= (t & 1u) << (wd * 4 + 0);
        bits |= ((t >> 8) & 1u) << (wd * 4 + 1);
        bits |= ((t >> 16) & 1u) << (wd * 4 + 2);
        bits |= ((t >> 24) & 1u) << (wd * 4 + 3);
      }
      mbits[mrl][mcw] = bits;
    }
    __syncthreads();

    v8f s[4];
#pragma unroll
    for (int nt = 0; nt < 4; ++nt) {
      const int key = nt * 16 + l16;
      v16bf b0 = load_bfrag(&k_lds[key][half * 16]);
      v16bf b1 = load_bfrag(&k_lds[key][32 + half * 16]);
      v8f acc = {};
      acc = wmma_bf16(qa0, b0, acc);
      acc = wmma_bf16(qa1, b1, acc);
      s[nt] = acc;
    }
#pragma unroll
    for (int r = 0; r < 8; ++r) {
      const int rl = wave * 16 + r + 8 * half;
      const unsigned w0 = mbits[rl][0], w1 = mbits[rl][1];
      s[0][r] = (w0 & bsel0) ? NEGV : s[0][r] * SCALE;
      s[1][r] = (w0 & bsel1) ? NEGV : s[1][r] * SCALE;
      s[2][r] = (w1 & bsel0) ? NEGV : s[2][r] * SCALE;
      s[3][r] = (w1 & bsel1) ? NEGV : s[3][r] * SCALE;
    }
    // normalized probabilities: write attn + stage for PV matmul
#pragma unroll
    for (int nt = 0; nt < 4; ++nt) {
#pragma unroll
      for (int r = 0; r < 8; ++r) {
        const int grow = row0 + wave * 16 + r + 8 * half;
        const int gcol = n0 + nt * 16 + l16;
        const float p = __expf(s[nt][r] - mrow[r]) * inv_l[r];
        ab[(size_t)grow * Nn + gcol] = p;
        qp.p[wave][r + 8 * half][nt * 16 + l16] = (__bf16)p;
      }
    }
    __syncthreads();   // make P visible across lanes before A-frag reads

    const v16bf pa0 = load_afrag(&qp.p[wave][l16][0], 0, half);
    const v16bf pa1 = load_afrag(&qp.p[wave][l16][0], 1, half);
#pragma unroll
    for (int dt = 0; dt < 4; ++dt) {
      const int dcol = dt * 16 + l16;
      v16bf vb0 = load_bfrag(&vt_lds[dcol][half * 16]);
      v16bf vb1 = load_bfrag(&vt_lds[dcol][32 + half * 16]);
      o[dt] = wmma_bf16(pa0, vb0, o[dt]);
      o[dt] = wmma_bf16(pa1, vb1, o[dt]);
    }
  }

  // ---- store O (fp32) ----
#pragma unroll
  for (int dt = 0; dt < 4; ++dt) {
#pragma unroll
    for (int r = 0; r < 8; ++r) {
      const int grow = row0 + wave * 16 + r + 8 * half;
      ob[(size_t)grow * Dd + dt * 16 + l16] = o[dt][r];
    }
  }
}

extern "C" void kernel_launch(void* const* d_in, const int* in_sizes, int n_in,
                              void* d_out, int out_size, void* d_ws, size_t ws_size,
                              hipStream_t stream) {
  const float* q = (const float*)d_in[0];
  const float* k = (const float*)d_in[1];
  const float* v = (const float*)d_in[2];
  const unsigned char* m = (const unsigned char*)d_in[3];
  float* out  = (float*)d_out;
  float* attn = out + (size_t)Bb * Hh * Nn * Dd;   // tuple: (out, attn) flat
  dim3 grid(Bb * Hh * (Nn / BM));
  attn_fwd_kernel<<<grid, 256, 0, stream>>>(q, k, v, m, out, attn);
}
